// CommunicationNetwork_30545807409620
// MI455X (gfx1250) — matrix operand order repaired and checked by reading, not compiled, not run on hardware
//
#include <hip/hip_runtime.h>

// ---------------------------------------------------------------------------
// CDNA5 (gfx1250) bf16 WMMA implementation of CommunicationNetwork forward.
// B=16384, S=37, F=17, OWN=9, H=256, E=128, NOUT=23
// ---------------------------------------------------------------------------

typedef __attribute__((ext_vector_type(16))) __bf16 v16bf;
typedef __attribute__((ext_vector_type(8)))  float  v8f;
typedef __attribute__((ext_vector_type(4)))  unsigned uint4v;

#define BATCH      16384
#define CAT_K      160            // 128 summed + 9 own + 23 zero pad (K padded to 32*5)
#define CAT_BYTES  ((long)BATCH * CAT_K * 2)

// packed-weight fragment offsets (units of fragments; 1 fragment = 256 dwords)
enum {
  OFF_S1 = 0,    // 17x256  -> 1x16  = 16
  OFF_S2 = 16,   // 256x256 -> 8x16  = 128
  OFF_S3 = 144,  // 256x128 -> 8x8   = 64
  OFF_A1 = 208,  // 137x256 -> 5x16  = 80
  OFF_A2 = 288,  // 256x256 -> 8x16  = 128
  OFF_A3 = 416,  // 256x23  -> 8x2   = 16
  OFF_V1 = 432,  // 137x256 -> 5x16  = 80
  OFF_V2 = 560,  // 256x256 -> 8x16  = 128
  OFF_V3 = 640,  // 256x1   -> 8x1   = 8
  FRAG_TOTAL = 648
};

// ---------------- helpers ----------------
static __device__ __forceinline__ unsigned short f2bf_rne(float f) {
  unsigned u = __builtin_bit_cast(unsigned, f);
  unsigned r = u + 0x7FFFu + ((u >> 16) & 1u);   // round-to-nearest-even (pack path)
  return (unsigned short)(r >> 16);
}
static __device__ __forceinline__ unsigned short f2bf_trunc(float f) {
  return (unsigned short)(__builtin_bit_cast(unsigned, f) >> 16);  // folds into d16_hi store
}
static __device__ __forceinline__ float bf2f(unsigned short h) {
  unsigned u = ((unsigned)h) << 16;
  return __builtin_bit_cast(float, u);
}
static __device__ __forceinline__ v8f splat8(float x) {
  v8f c;
#pragma unroll
  for (int i = 0; i < 8; ++i) c[i] = x;
  return c;
}
// ReLU as a single v_med3_f32 (no NaN-canonicalization double-max).
static __device__ __forceinline__ float relu1(float x) {
  return __builtin_amdgcn_fmed3f(x, 0.f, __builtin_inff());
}

static __device__ __forceinline__ v8f wmma_bf16(v16bf a, v16bf b, v8f c) {
  return __builtin_amdgcn_wmma_f32_16x16x32_bf16(
      /*neg_a=*/false, a, /*neg_b=*/false, b,
      /*c_mod=*/(short)0, c, /*reuse_a=*/false, /*reuse_b=*/false);
}

// B fragment: packed layout, 32 bytes per lane, contiguous -> 2x b128 loads.
static __device__ __forceinline__ v16bf load_bfrag(const unsigned* fragbase, int lane) {
  union { uint4v q[2]; v16bf v; } t;
  const uint4v* p = (const uint4v*)fragbase + lane * 2;
  t.q[0] = p[0];
  t.q[1] = p[1];
  return t.v;
}

// A fragment (16x32 bf16, MxK) from row-major bf16 activations.
// Lane L: g=L/16, m=mbase+L%16. Per-lane data is two contiguous 16B runs:
// k = kbase+8g .. +7 (dwords d0..d3) and k = kbase+16+8g .. +7 (d4..d7),
// matching ISA 7.12.2 "16-bit A-Matrix 16x32". -> two b128 loads per lane.
static __device__ __forceinline__ v16bf load_afrag(const unsigned short* act,
                                                   int stride, long mbase,
                                                   int lane, int kbase) {
  const int g = lane >> 4;
  const long m = mbase + (lane & 15);
  const unsigned short* row = act + m * (long)stride + kbase + 8 * g;
  union { uint4v q[2]; v16bf v; } t;
  t.q[0] = *(const uint4v*)(row);
  t.q[1] = *(const uint4v*)(row + 16);
  return t.v;
}

// C/D tile (16x16 f32): lane L -> n = nt*16 + L%16; VGPR r -> m = r + 8*(L/16).
// Bias is pre-loaded into the accumulator, so epilogue = med3 + optional mask +
// truncating bf16 store (ds_store_b16_d16_hi, no conversion ALU).
static __device__ __forceinline__ void store_tile(unsigned short* out, int stride,
                                                  int mbase, int nt, int lane,
                                                  v8f c, const float* mask) {
  const int g = lane >> 4;
  const int n = nt * 16 + (lane & 15);
#pragma unroll
  for (int r = 0; r < 8; ++r) {
    int m = mbase + r + 8 * g;
    float v = relu1(c[r]);
    if (mask) v *= mask[m];
    out[m * stride + n] = f2bf_trunc(v);
  }
}

// ---------------- kernel 0: weight packing ----------------
// One block per fragment, 256 threads: thread -> (lane = tid/8, d = tid%8).
// dword holds bf16(W[k0][n]) | bf16(W[k0+1][n])<<16, k0 = kt*32 + 2(d&3) + 8g + 16(d>>2).
__global__ void pack_weight_kernel(const float* __restrict__ W, unsigned* __restrict__ dst,
                                   int K, int N, int NT) {
  const int f = blockIdx.x;
  const int tid = threadIdx.x;
  const int lane = tid >> 3, d = tid & 7;
  const int kt = f / NT, nt = f % NT;
  const int g = lane >> 4;
  const int n = nt * 16 + (lane & 15);
  const int k0 = kt * 32 + 2 * (d & 3) + 8 * g + 16 * (d >> 2);
  float lo = (k0 < K && n < N) ? W[(long)k0 * N + n] : 0.f;
  float hi = (k0 + 1 < K && n < N) ? W[(long)(k0 + 1) * N + n] : 0.f;
  dst[(long)f * 256 + tid] = (unsigned)f2bf_rne(lo) | ((unsigned)f2bf_rne(hi) << 16);
}

// ---------------- kernel 1: sequence encoder ----------------
// One block (128 threads, 4 waves) per batch element.
// obs[b] async-copied to LDS, then relu MLP 17->256->256->128, masked sum.
#define IN_STRIDE 40     // 32 cols padded to 40 (80B rows, 16B aligned)
#define H_STRIDE  264    // 256 cols padded to 264 (528B rows, 16B aligned)
#define OBS_ROW   646    // 38*17 f32 per batch element

__global__ void __launch_bounds__(128)
encoder_kernel(const float* __restrict__ obs, const unsigned* __restrict__ packW,
               const float* __restrict__ sb1, const float* __restrict__ sb2,
               const float* __restrict__ sb3, unsigned short* __restrict__ cat) {
  __shared__ __align__(16) float          sStage[OBS_ROW + 2];
  __shared__ __align__(16) unsigned short sIn[48 * IN_STRIDE];
  __shared__               float          sMask[48];
  __shared__ __align__(16) unsigned short sB1[48 * H_STRIDE];
  __shared__ __align__(16) unsigned short sB2[48 * H_STRIDE];

  const int b    = blockIdx.x;
  const int tid  = threadIdx.x;
  const int wv   = tid >> 5;
  const int lane = tid & 31;
  const float* ob = obs + (long)b * OBS_ROW;

  // Async-copy the whole obs[b] slab (646 dwords) into LDS (ASYNCcnt path).
  for (int i = tid; i < OBS_ROW; i += 128) {
    unsigned lds_off = (unsigned)(size_t)&sStage[i];
    unsigned long long gaddr = (unsigned long long)(size_t)(ob + i);
    asm volatile("global_load_async_to_lds_b32 %0, %1, off"
                 :: "v"(lds_off), "v"(gaddr) : "memory");
  }
  asm volatile("s_wait_asynccnt 0x0" ::: "memory");
  __syncthreads();

  // bf16 A-tile staging (48x32, zero padded) + per-row mask from f32 data.
  for (int i = tid; i < 48 * 32; i += 128) {
    int s = i >> 5, f = i & 31;
    float v = (s < 37 && f < 17) ? sStage[(1 + s) * 17 + f] : 0.f;
    sIn[s * IN_STRIDE + f] = f2bf_trunc(v);
  }
  if (tid < 48) {
    float acc = 0.f;
    if (tid < 37)
      for (int f = 0; f < 17; ++f) acc += fabsf(sStage[(1 + tid) * 17 + f]);
    sMask[tid] = (acc != 0.f) ? 1.f : 0.f;
  }
  __syncthreads();

  // Layer 1: (48x32) x (32x256), KT=1; each wave handles nt = wv + 4j (uniform).
  for (int mt = 0; mt < 3; ++mt) {
    v16bf a = load_afrag(sIn, IN_STRIDE, mt * 16, lane, 0);
#pragma unroll
    for (int j = 0; j < 4; ++j) {
      const int nt = wv + 4 * j;
      v8f c = splat8(sb1[nt * 16 + (lane & 15)]);
      c = wmma_bf16(a, load_bfrag(packW + (OFF_S1 + nt) * 256, lane), c);
      store_tile(sB1, H_STRIDE, mt * 16, nt, lane, c, nullptr);
    }
  }
  __syncthreads();

  // Layer 2: (48x256) x (256x256), KT=8
  for (int mt = 0; mt < 3; ++mt) {
    v16bf a[8];
#pragma unroll
    for (int kt = 0; kt < 8; ++kt) a[kt] = load_afrag(sB1, H_STRIDE, mt * 16, lane, kt * 32);
#pragma unroll
    for (int j = 0; j < 4; ++j) {
      const int nt = wv + 4 * j;
      v8f c = splat8(sb2[nt * 16 + (lane & 15)]);
#pragma unroll
      for (int kt = 0; kt < 8; ++kt)
        c = wmma_bf16(a[kt], load_bfrag(packW + (OFF_S2 + kt * 16 + nt) * 256, lane), c);
      store_tile(sB2, H_STRIDE, mt * 16, nt, lane, c, nullptr);
    }
  }
  __syncthreads();

  // Layer 3: (48x256) x (256x128), KT=8, masked; write into sB1 (reuse)
  for (int mt = 0; mt < 3; ++mt) {
    v16bf a[8];
#pragma unroll
    for (int kt = 0; kt < 8; ++kt) a[kt] = load_afrag(sB2, H_STRIDE, mt * 16, lane, kt * 32);
#pragma unroll
    for (int j = 0; j < 2; ++j) {
      const int nt = wv + 4 * j;
      v8f c = splat8(sb3[nt * 16 + (lane & 15)]);
#pragma unroll
      for (int kt = 0; kt < 8; ++kt)
        c = wmma_bf16(a[kt], load_bfrag(packW + (OFF_S3 + kt * 8 + nt) * 256, lane), c);
      store_tile(sB1, H_STRIDE, mt * 16, nt, lane, c, sMask);
    }
  }
  __syncthreads();

  // Masked sum over rows -> cat[b][0:128]; own features -> cat[b][128:137]; pad 0.
  {
    float acc = 0.f;
    for (int m = 0; m < 48; ++m) acc += bf2f(sB1[m * H_STRIDE + tid]);
    cat[(long)b * CAT_K + tid] = f2bf_trunc(acc);
  }
  if (tid < 32) {
    float v = (tid < 9) ? sStage[tid] : 0.f;   // obs[b][0][tid]
    cat[(long)b * CAT_K + 128 + tid] = f2bf_trunc(v);
  }
}

// ---------------- kernel 2: actor / value heads ----------------
// 128 threads = 4 waves; each wave owns 16 rows. Per-wave LDS ping-pong,
// no cross-wave sharing -> no barriers.
__global__ void __launch_bounds__(128)
head_kernel(const unsigned short* __restrict__ cat, const unsigned* __restrict__ packW,
            const float* __restrict__ ab1, const float* __restrict__ ab2,
            const float* __restrict__ ab3, const float* __restrict__ vb1,
            const float* __restrict__ vb2, const float* __restrict__ vb3,
            float* __restrict__ out) {
  __shared__ __align__(16) unsigned short sh[4 * 2 * 16 * H_STRIDE];

  const int tid  = threadIdx.x;
  const int wv   = tid >> 5;
  const int lane = tid & 31;
  const int g    = lane >> 4;
  const long rbase = ((long)blockIdx.x * 4 + wv) * 16;

  unsigned short* h1 = sh + wv * (2 * 16 * H_STRIDE);
  unsigned short* h2 = h1 + 16 * H_STRIDE;

  for (int br = 0; br < 2; ++br) {
    const unsigned* W1 = packW + (br ? OFF_V1 : OFF_A1) * 256;
    const unsigned* W2 = packW + (br ? OFF_V2 : OFF_A2) * 256;
    const unsigned* W3 = packW + (br ? OFF_V3 : OFF_A3) * 256;
    const float* b1 = br ? vb1 : ab1;
    const float* b2 = br ? vb2 : ab2;

    v16bf a[8];

    // L1: cat (16x160) x (160x256), KT=5; A-frags straight from global
#pragma unroll
    for (int kt = 0; kt < 5; ++kt) a[kt] = load_afrag(cat, CAT_K, rbase, lane, kt * 32);
#pragma unroll 4
    for (int nt = 0; nt < 16; ++nt) {
      v8f c = splat8(b1[nt * 16 + (lane & 15)]);
#pragma unroll
      for (int kt = 0; kt < 5; ++kt)
        c = wmma_bf16(a[kt], load_bfrag(W1 + (kt * 16 + nt) * 256, lane), c);
      store_tile(h1, H_STRIDE, 0, nt, lane, c, nullptr);
    }

    // L2: (16x256) x (256x256), KT=8
#pragma unroll
    for (int kt = 0; kt < 8; ++kt) a[kt] = load_afrag(h1, H_STRIDE, 0, lane, kt * 32);
#pragma unroll 4
    for (int nt = 0; nt < 16; ++nt) {
      v8f c = splat8(b2[nt * 16 + (lane & 15)]);
#pragma unroll
      for (int kt = 0; kt < 8; ++kt)
        c = wmma_bf16(a[kt], load_bfrag(W2 + (kt * 16 + nt) * 256, lane), c);
      store_tile(h2, H_STRIDE, 0, nt, lane, c, nullptr);
    }

    // L3
#pragma unroll
    for (int kt = 0; kt < 8; ++kt) a[kt] = load_afrag(h2, H_STRIDE, 0, lane, kt * 32);

    if (br == 0) {
      // actor logits: (16x256) x (256x23), NT=2 (N padded to 32)
#pragma unroll
      for (int nt = 0; nt < 2; ++nt) {
        const int n = nt * 16 + (lane & 15);
        v8f c = splat8((n < 23) ? ab3[n] : 0.f);
#pragma unroll
        for (int kt = 0; kt < 8; ++kt)
          c = wmma_bf16(a[kt], load_bfrag(W3 + (kt * 2 + nt) * 256, lane), c);
        if (n < 23) {
#pragma unroll
          for (int r = 0; r < 8; ++r) {
            long row = rbase + r + 8 * g;
            out[row * 23 + n] = c[r];
          }
        }
      }
    } else {
      // value: (16x256) x (256x1), NT=1
      v8f c = splat8(vb3[0]);
#pragma unroll
      for (int kt = 0; kt < 8; ++kt)
        c = wmma_bf16(a[kt], load_bfrag(W3 + kt * 256, lane), c);
      if ((lane & 15) == 0) {
#pragma unroll
        for (int r = 0; r < 8; ++r) {
          long row = rbase + r + 8 * g;
          out[(long)BATCH * 23 + row] = c[r];
        }
      }
    }
  }
}

// ---------------- host entry ----------------
extern "C" void kernel_launch(void* const* d_in, const int* in_sizes, int n_in,
                              void* d_out, int out_size, void* d_ws, size_t ws_size,
                              hipStream_t stream) {
  const float* obs = (const float*)d_in[0];
  const float* sW1 = (const float*)d_in[1];  const float* sb1 = (const float*)d_in[2];
  const float* sW2 = (const float*)d_in[3];  const float* sb2 = (const float*)d_in[4];
  const float* sW3 = (const float*)d_in[5];  const float* sb3 = (const float*)d_in[6];
  const float* aW1 = (const float*)d_in[7];  const float* ab1 = (const float*)d_in[8];
  const float* aW2 = (const float*)d_in[9];  const float* ab2 = (const float*)d_in[10];
  const float* aW3 = (const float*)d_in[11]; const float* ab3 = (const float*)d_in[12];
  const float* vW1 = (const float*)d_in[13]; const float* vb1 = (const float*)d_in[14];
  const float* vW2 = (const float*)d_in[15]; const float* vb2 = (const float*)d_in[16];
  const float* vW3 = (const float*)d_in[17]; const float* vb3 = (const float*)d_in[18];

  unsigned short* cat = (unsigned short*)d_ws;
  unsigned* packW = (unsigned*)((char*)d_ws + CAT_BYTES);

  // Pack all weights into bf16 WMMA B-fragment order (grid = fragment count).
  pack_weight_kernel<<< 16, 256, 0, stream>>>(sW1, packW + OFF_S1 * 256,  17, 256, 16);
  pack_weight_kernel<<<128, 256, 0, stream>>>(sW2, packW + OFF_S2 * 256, 256, 256, 16);
  pack_weight_kernel<<< 64, 256, 0, stream>>>(sW3, packW + OFF_S3 * 256, 256, 128,  8);
  pack_weight_kernel<<< 80, 256, 0, stream>>>(aW1, packW + OFF_A1 * 256, 137, 256, 16);
  pack_weight_kernel<<<128, 256, 0, stream>>>(aW2, packW + OFF_A2 * 256, 256, 256, 16);
  pack_weight_kernel<<< 16, 256, 0, stream>>>(aW3, packW + OFF_A3 * 256, 256,  23,  2);
  pack_weight_kernel<<< 80, 256, 0, stream>>>(vW1, packW + OFF_V1 * 256, 137, 256, 16);
  pack_weight_kernel<<<128, 256, 0, stream>>>(vW2, packW + OFF_V2 * 256, 256, 256, 16);
  pack_weight_kernel<<<  8, 256, 0, stream>>>(vW3, packW + OFF_V3 * 256, 256,   1,  1);

  encoder_kernel<<<BATCH, 128, 0, stream>>>(obs, packW, sb1, sb2, sb3, cat);
  head_kernel<<<BATCH / 64, 128, 0, stream>>>(cat, packW, ab1, ab2, ab3, vb1, vb2, vb3,
                                              (float*)d_out);
}
